// SwinTransformer_53841710023219
// MI455X (gfx1250) — matrix operand order repaired and checked by reading, not compiled, not run on hardware
//
#include <hip/hip_runtime.h>

#define DCH    512
#define NHEADS 8
#define HDIM   64
#define TOK    49
#define TPAD   64
#define WIN    7

typedef __attribute__((ext_vector_type(16))) _Float16 v16h;
typedef __attribute__((ext_vector_type(8)))  _Float16 v8h;
typedef __attribute__((ext_vector_type(4)))  _Float16 v4h;
typedef __attribute__((ext_vector_type(8)))  float    v8f;
typedef __attribute__((ext_vector_type(4)))  float    v4f;

// ---------------------------------------------------------------------------
// B-fragment flat index for a 512x512 (K x N) f16 matrix, WMMA 16x16x32:
//   tile (kt, ntile); lane L = hg*16 + n holds 16 contiguous halves, element e
//   maps to K = kt*32 + 16*hg + e, N = ntile*16 + n.
// ---------------------------------------------------------------------------
__device__ __forceinline__ int wpk_flat(int din, int dout) {
    int kt = din >> 5, hg = (din >> 4) & 1, e = din & 15;
    int nt = dout >> 4, n = dout & 15;
    return (((kt * 32 + nt) * 32) + hg * 16 + n) * 16 + e;
}

__global__ void pack_weights(const float* __restrict__ wq,
                             const float* __restrict__ wk,
                             const float* __restrict__ wv,
                             _Float16* __restrict__ wpk) {
    int g = blockIdx.x * blockDim.x + threadIdx.x;
    const int per = DCH * DCH;
    if (g >= 3 * per) return;
    int w = g / per, rem = g - w * per;
    int din = rem / DCH, dout = rem - din * DCH;
    const float* src = (w == 0) ? wq : (w == 1) ? wk : wv;
    wpk[(size_t)w * per + wpk_flat(din, dout)] = (_Float16)src[rem];
}

__device__ __forceinline__ v16h cat16(v8h lo, v8h hi) {
    return __builtin_shufflevector(lo, hi, 0, 1, 2, 3, 4, 5, 6, 7,
                                           8, 9, 10, 11, 12, 13, 14, 15);
}

// A-fragment (16x32 f16, contiguous k order): lane m = lane&15, hg = lane>>4.
// e=0..15 -> K = k0 + 16*hg + e (two ds_read_b128)
__device__ __forceinline__ v16h load_a(const _Float16* p) {
    v8h lo = *(const v8h*)p;
    v8h hi = *(const v8h*)(p + 8);
    return cat16(lo, hi);
}

__device__ __forceinline__ v8f wmma_f16(v16h a, v16h b, v8f c) {
    return __builtin_amdgcn_wmma_f32_16x16x32_f16(false, a, false, b,
                                                  (short)0, c, false, false);
}

__launch_bounds__(256, 1)
__global__ void swin_attn(const float* __restrict__ x,
                          const _Float16* __restrict__ wpk,
                          const float* __restrict__ bq,
                          const float* __restrict__ bk,
                          const float* __restrict__ bv,
                          const float* __restrict__ pos_bias,
                          float* __restrict__ out) {
    // 4 x (64 x 512) f16 = 256 KB.
    //   [0]      xs: f16 window (phase1 A); probs region in phase2
    //   [1],[2]  qs, ks row-major; ALSO the fp32 async staging area in stage 0
    //   [3]      vsB: V packed in B-fragment order
    __shared__ __align__(32) _Float16 smem[4 * TPAD * DCH];
    _Float16* xs  = smem;
    _Float16* qs  = smem + 1 * TPAD * DCH;
    _Float16* ks  = smem + 2 * TPAD * DCH;
    _Float16* vsB = smem + 3 * TPAD * DCH;

    const int tid  = threadIdx.x;
    const int lane = tid & 31;
    const int wave = tid >> 5;
    const int n    = lane & 15;   // N-within-tile / M-within-tile
    const int hg   = lane >> 4;   // lane half-group

    const int bid = blockIdx.x;   // b*64 + window
    const int b   = bid >> 6;
    const int nw  = bid & 63;
    const int wy  = nw >> 3, wx = nw & 7;

    // ---- Stage 0a: async global->LDS copy of the fp32 window --------------
    // 49 rows x 2048 B staged at LDS byte offset 65536 (qs/ks area).
    const unsigned stage_base = (unsigned)(TPAD * DCH * sizeof(_Float16)); // 65536
    for (int i = tid; i < TOK * 128; i += 256) {          // 128 x b128 per row
        int row   = i >> 7;
        int col16 = (i & 127) * 16;                        // byte offset in row
        int ty = row / WIN, tx = row - ty * WIN;
        const char* g = (const char*)(x + ((size_t)(b * 56 + wy * WIN + ty) * 56
                                           + (wx * WIN + tx)) * DCH) + col16;
        unsigned ldsa = stage_base + (unsigned)(row * 2048 + col16);
        asm volatile("global_load_async_to_lds_b128 %0, %1, off"
                     :: "v"(ldsa), "v"(g) : "memory");
    }
    asm volatile("s_wait_asynccnt 0x0" ::: "memory");
    __syncthreads();

    // ---- Stage 0b: fp32 staging -> f16 xs, zero-pad rows 49..63 -----------
    const float* stagef = (const float*)((const char*)smem + stage_base);
    for (int v = tid; v < TPAD * (DCH / 4); v += 256) {
        int row = v >> 7;
        int c4  = (v & 127) * 4;
        v4h h4;
        if (row < TOK) {
            v4f f = *(const v4f*)(stagef + row * DCH + c4);
            h4[0] = (_Float16)f[0]; h4[1] = (_Float16)f[1];
            h4[2] = (_Float16)f[2]; h4[3] = (_Float16)f[3];
        } else {
            h4[0] = h4[1] = h4[2] = h4[3] = (_Float16)0.0f;
        }
        *(v4h*)&xs[row * DCH + c4] = h4;
    }
    __syncthreads();   // staging dead; qs/ks may now be written

    // ---- Stage 1: Q/K/V = win * W + b; each wave owns 4 N-tile columns ----
    // B-fragment loaded once per (kt,nt) and reused across all 4 M-tiles:
    // per-block weight reads = 1.5 MB (the minimum), 4 WMMAs per B-load.
    const float invs = 0.044194173824159216f; // 1/sqrt(512)
    for (int m3 = 0; m3 < 3; ++m3) {
        const _Float16* wp = wpk + (size_t)m3 * DCH * DCH;
        const float* bias  = (m3 == 0) ? bq : (m3 == 1) ? bk : bv;
        for (int c = 0; c < 4; ++c) {
            int nt = wave * 4 + c;
            v8f acc[4] = {};
            for (int kt = 0; kt < 16; ++kt) {
                v16h bf = *(const v16h*)(wp + (((kt * 32 + nt) * 32) + lane) * 16);
                #pragma unroll
                for (int mt = 0; mt < 4; ++mt) {
                    v16h a = load_a(xs + (mt * 16 + n) * DCH + kt * 32 + 16 * hg);
                    acc[mt] = wmma_f16(a, bf, acc[mt]);
                }
            }
            float bc = bias[nt * 16 + n];
            #pragma unroll
            for (int mt = 0; mt < 4; ++mt) {
                if (m3 == 0) {
                    #pragma unroll
                    for (int r = 0; r < 8; ++r)
                        qs[(mt * 16 + 8 * hg + r) * DCH + nt * 16 + n] =
                            (_Float16)((acc[mt][r] + bc) * invs);
                } else if (m3 == 1) {
                    #pragma unroll
                    for (int r = 0; r < 8; ++r)
                        ks[(mt * 16 + 8 * hg + r) * DCH + nt * 16 + n] =
                            (_Float16)(acc[mt][r] + bc);
                } else {
                    #pragma unroll
                    for (int r = 0; r < 8; ++r) {
                        int M = mt * 16 + 8 * hg + r;   // token (K dim of P*V)
                        int kt2 = M >> 5, hg2 = (M >> 4) & 1, e = M & 15;
                        vsB[(((kt2 * 32 + nt) * 32) + hg2 * 16 + n) * 16 + e] =
                            (_Float16)(acc[mt][r] + bc);
                    }
                }
            }
        }
    }
    __syncthreads();   // Q/K/V visible to all waves; xs free for probs

    // ---- Stage 2: per-head attention (wave == head) -----------------------
    const int h = wave;
    _Float16* probs = xs + h * (TPAD * TPAD);   // 64x64 f16 per head

    for (int it = 0; it < 4; ++it) {
        // Q fragments for this M-tile, reused across all 4 jt tiles
        v16h a0 = load_a(qs + (it * 16 + n) * DCH + h * HDIM + 16 * hg);
        v16h a1 = load_a(qs + (it * 16 + n) * DCH + h * HDIM + 32 + 16 * hg);
        v8f sacc[4];
        #pragma unroll
        for (int jt = 0; jt < 4; ++jt) {
            // B = K^T fragment: contiguous row read from row-major K
            v16h b0 = *(const v16h*)(ks + (jt * 16 + n) * DCH + h * HDIM
                                     + 16 * hg);
            v16h b1 = *(const v16h*)(ks + (jt * 16 + n) * DCH + h * HDIM
                                     + 32 + 16 * hg);
            v8f acc = {};
            acc = wmma_f16(a0, b0, acc);
            acc = wmma_f16(a1, b1, acc);
            int col = jt * 16 + n;
            #pragma unroll
            for (int r = 0; r < 8; ++r) {
                int row = it * 16 + 8 * hg + r;
                float v = acc[r];
                if (row < TOK && col < TOK)
                    v += pos_bias[h * TOK * TOK + row * TOK + col];
                if (col >= TOK || col == row) v = -3.0e38f;  // pad + eye mask
                acc[r] = v;
            }
            sacc[jt] = acc;
        }
        // softmax over 64 cols: 4 regs + 16-lane shuffle reduction (wave32)
        #pragma unroll
        for (int r = 0; r < 8; ++r) {
            float m = fmaxf(fmaxf(sacc[0][r], sacc[1][r]),
                            fmaxf(sacc[2][r], sacc[3][r]));
            for (int msk = 8; msk >= 1; msk >>= 1)
                m = fmaxf(m, __shfl_xor(m, msk, 32));
            float e0 = __expf(sacc[0][r] - m);
            float e1 = __expf(sacc[1][r] - m);
            float e2 = __expf(sacc[2][r] - m);
            float e3 = __expf(sacc[3][r] - m);
            float s = e0 + e1 + e2 + e3;
            for (int msk = 8; msk >= 1; msk >>= 1)
                s += __shfl_xor(s, msk, 32);
            float inv = 1.0f / s;
            int row = it * 16 + 8 * hg + r;
            probs[row * TPAD +  0 + n] = (_Float16)(e0 * inv);
            probs[row * TPAD + 16 + n] = (_Float16)(e1 * inv);
            probs[row * TPAD + 32 + n] = (_Float16)(e2 * inv);
            probs[row * TPAD + 48 + n] = (_Float16)(e3 * inv);
        }
    }

    // ---- Stage 3: O = P * V_h; V fragments hoisted across M-tiles ---------
    for (int nt2 = 0; nt2 < 4; ++nt2) {
        v16h b0 = *(const v16h*)(vsB + ((( 0 + (h * 4 + nt2)) * 32) + lane) * 16);
        v16h b1 = *(const v16h*)(vsB + (((32 + (h * 4 + nt2)) * 32) + lane) * 16);
        #pragma unroll
        for (int it = 0; it < 4; ++it) {
            v8f acc = {};
            acc = wmma_f16(load_a(probs + (it * 16 + n) * TPAD + 16 * hg),
                           b0, acc);
            acc = wmma_f16(load_a(probs + (it * 16 + n) * TPAD + 32 + 16 * hg),
                           b1, acc);
            #pragma unroll
            for (int r = 0; r < 8; ++r) {
                int token = it * 16 + 8 * hg + r;
                if (token < TOK)
                    out[((size_t)bid * TOK + token) * DCH
                        + h * HDIM + nt2 * 16 + n] = acc[r];
            }
        }
    }
}

extern "C" void kernel_launch(void* const* d_in, const int* in_sizes, int n_in,
                              void* d_out, int out_size, void* d_ws, size_t ws_size,
                              hipStream_t stream) {
    (void)in_sizes; (void)n_in; (void)out_size; (void)ws_size;
    const float* x  = (const float*)d_in[0];
    const float* wq = (const float*)d_in[1];
    const float* bq = (const float*)d_in[2];
    const float* wk = (const float*)d_in[3];
    const float* bk = (const float*)d_in[4];
    const float* wv = (const float*)d_in[5];
    const float* bv = (const float*)d_in[6];
    const float* pb = (const float*)d_in[7];
    _Float16* wpk = (_Float16*)d_ws;   // 3 * 512*512 * 2B = 1.5 MB scratch

    int total = 3 * DCH * DCH;
    pack_weights<<<(total + 255) / 256, 256, 0, stream>>>(wq, wk, wv, wpk);
    swin_attn<<<32 * 64, 256, 0, stream>>>(x, wpk, bq, bk, bv, pb, (float*)d_out);
}
